// Attention_81260781240960
// MI455X (gfx1250) — matrix operand order repaired and checked by reading, not compiled, not run on hardware
//
#include <hip/hip_runtime.h>

// ---------------------------------------------------------------------------
// Cross-attention (B=8, N=M=1024, D=512, H=8, dh=64) on gfx1250 WMMA (bf16),
// with TDM (tensor_load_to_lds) staging of K/V tiles and permlane16 softmax
// reductions.
// ---------------------------------------------------------------------------

typedef __bf16 bf16;
typedef __attribute__((ext_vector_type(4)))  bf16  v4bf;
typedef __attribute__((ext_vector_type(8)))  bf16  v8bf;
typedef __attribute__((ext_vector_type(16))) bf16  v16bf;
typedef __attribute__((ext_vector_type(8)))  float v8f;
typedef __attribute__((ext_vector_type(4)))  unsigned int v4u;
typedef __attribute__((ext_vector_type(8)))  int   v8i;
typedef __attribute__((ext_vector_type(4)))  int   v4i;

#if defined(__has_builtin)
#if __has_builtin(__builtin_amdgcn_tensor_load_to_lds) && \
    __has_builtin(__builtin_amdgcn_s_wait_tensorcnt)
#define USE_TDM 1
#endif
#if __has_builtin(__builtin_amdgcn_permlane16)
#define HAVE_PERMLANE16 1
#endif
#endif
#ifndef USE_TDM
#define USE_TDM 0
#endif
#ifndef HAVE_PERMLANE16
#define HAVE_PERMLANE16 0
#endif

#define DEV_INLINE __device__ __forceinline__

DEV_INLINE bf16 to_bf16(float x) { return (bf16)x; }
DEV_INLINE bf16 to_bf16(bf16 x)  { return x; }

DEV_INLINE v8bf ld8(const bf16* p) { return *(const v8bf*)p; }

DEV_INLINE v16bf combine16(v8bf lo, v8bf hi) {
  v16bf r;
#pragma unroll
  for (int i = 0; i < 8; ++i) { r[i] = lo[i]; r[i + 8] = hi[i]; }
  return r;
}

DEV_INLINE v8f wmma_bf16(v16bf a, v16bf b, v8f c) {
  // (neg_a, A, neg_b, B, c_mod, C, reuse_a, reuse_b)
  return __builtin_amdgcn_wmma_f32_16x16x32_bf16(false, a, false, b,
                                                 (short)0, c, false, false);
}

// ---- 16-lane-row butterfly reductions (VALU permlane16, no DS waits) ------
#if HAVE_PERMLANE16
DEV_INLINE float permxor16(float x, unsigned sel_lo, unsigned sel_hi) {
  unsigned u = __builtin_bit_cast(unsigned, x);
  u = __builtin_amdgcn_permlane16(u, u, sel_lo, sel_hi, false, false);
  return __builtin_bit_cast(float, u);
}
DEV_INLINE float redmax16(float x) {
  x = fmaxf(x, permxor16(x, 0x67452301u, 0xEFCDAB89u));  // xor 1
  x = fmaxf(x, permxor16(x, 0x54761032u, 0xDCFE98BAu));  // xor 2
  x = fmaxf(x, permxor16(x, 0x32107654u, 0xBA98FEDCu));  // xor 4
  x = fmaxf(x, permxor16(x, 0xFEDCBA98u, 0x76543210u));  // xor 8
  return x;
}
DEV_INLINE float redsum16(float x) {
  x += permxor16(x, 0x67452301u, 0xEFCDAB89u);
  x += permxor16(x, 0x54761032u, 0xDCFE98BAu);
  x += permxor16(x, 0x32107654u, 0xBA98FEDCu);
  x += permxor16(x, 0xFEDCBA98u, 0x76543210u);
  return x;
}
#else
DEV_INLINE float redmax16(float x) {
#pragma unroll
  for (int m = 1; m < 16; m <<= 1) x = fmaxf(x, __shfl_xor(x, m, 32));
  return x;
}
DEV_INLINE float redsum16(float x) {
#pragma unroll
  for (int m = 1; m < 16; m <<= 1) x += __shfl_xor(x, m, 32);
  return x;
}
#endif

// ---- TDM descriptor issue (2D tile, bf16, LDS row padding enabled) --------
#if USE_TDM
DEV_INLINE void tdm_load(unsigned lds_off, const bf16* gptr,
                         unsigned d0, unsigned d1, unsigned d2, unsigned d3,
                         unsigned d4, unsigned d5) {
  unsigned long long ga = (unsigned long long)gptr;
  // D# group 0: count=1; lds_addr; global_addr[56:0]; type=2 at [127:126]
  v4u g0 = {1u, lds_off, (unsigned)ga,
            (unsigned)((ga >> 32) & 0x01FFFFFFu) | 0x80000000u};
  // D# group 1: data_size/pad in d0; tensor dims, tile dims, stride in d1..d5
  v8i g1 = {(int)d0, (int)d1, (int)d2, (int)d3, (int)d4, (int)d5, 0, 0};
  v4i z4 = {0, 0, 0, 0};
#if __clang_major__ >= 23
  v8i z8 = {0, 0, 0, 0, 0, 0, 0, 0};
  __builtin_amdgcn_tensor_load_to_lds(g0, g1, z4, z4, z8, 0);
#else
  __builtin_amdgcn_tensor_load_to_lds(g0, g1, z4, z4, 0);
#endif
}
#endif

enum { MODE_Q = 0, MODE_KV = 1, MODE_OUT = 2 };

// ---------------------------------------------------------------------------
// Generic 128x128-tile bf16 WMMA GEMM:  C[R, Cdim] = A[R, Kdim] * W[Kdim, Cdim]
// 256 threads = 8 waves (2x4); each wave computes 4x2 16x16 WMMA tiles.
// ---------------------------------------------------------------------------
template <int MODE, typename AT>
__global__ __launch_bounds__(256) void gemm_wmma_kernel(
    const AT* __restrict__ A, const float* __restrict__ W,
    const float* __restrict__ bias,
    bf16* __restrict__ oq, bf16* __restrict__ ok, bf16* __restrict__ ov,
    float* __restrict__ of, int Kdim, int Cdim) {
  constexpr int LDA = 40;  // bf16 elems per LDS row (80 B, 16B-aligned)
  __shared__ bf16 As[128 * LDA];
  __shared__ bf16 Bs[128 * LDA];  // transposed: [n][k]

  const int tid  = threadIdx.x;
  const int wave = tid >> 5, lane = tid & 31;
  const int wr = wave >> 2, wc = wave & 3;
  const int half = lane >> 4, l16 = lane & 15;
  const int rowBase = blockIdx.x * 128;
  const int colBase = blockIdx.y * 128;

  v8f acc[4][2] = {};

  for (int k0 = 0; k0 < Kdim; k0 += 32) {
#pragma unroll
    for (int i = 0; i < 4; ++i) {
      int lin = i * 256 + tid;
      int r   = lin >> 3;
      int kq  = (lin & 7) << 2;
      const AT* src = A + (size_t)(rowBase + r) * Kdim + k0 + kq;
      v4bf pk;
#pragma unroll
      for (int j = 0; j < 4; ++j) pk[j] = to_bf16(src[j]);
      *(v4bf*)&As[r * LDA + kq] = pk;
    }
#pragma unroll
    for (int i = 0; i < 4; ++i) {
      int lin = i * 256 + tid;
      int kr  = lin >> 5;
      int cq  = (lin & 31) << 2;
      const float* src = W + (size_t)(k0 + kr) * Cdim + colBase + cq;
#pragma unroll
      for (int j = 0; j < 4; ++j) Bs[(cq + j) * LDA + kr] = to_bf16(src[j]);
    }
    __syncthreads();

    v16bf af[4];
#pragma unroll
    for (int mt = 0; mt < 4; ++mt) {
      const bf16* p = &As[(wr * 64 + mt * 16 + l16) * LDA + half * 8];
      af[mt] = combine16(ld8(p), ld8(p + 16));
    }
    v16bf bfr[2];
#pragma unroll
    for (int nt = 0; nt < 2; ++nt) {
      const bf16* p = &Bs[(wc * 32 + nt * 16 + l16) * LDA + half * 8];
      bfr[nt] = combine16(ld8(p), ld8(p + 16));
    }
#pragma unroll
    for (int mt = 0; mt < 4; ++mt)
#pragma unroll
      for (int nt = 0; nt < 2; ++nt)
        acc[mt][nt] = wmma_bf16(af[mt], bfr[nt], acc[mt][nt]);
    __syncthreads();
  }

#pragma unroll
  for (int mt = 0; mt < 4; ++mt) {
#pragma unroll
    for (int nt = 0; nt < 2; ++nt) {
      int gr0 = rowBase + wr * 64 + mt * 16 + half * 8;
      int gc  = colBase + wc * 32 + nt * 16 + l16;
#pragma unroll
      for (int r = 0; r < 8; ++r) {
        float v = acc[mt][nt][r];
        int row = gr0 + r;
        if (MODE == MODE_Q) {
          int b = row >> 10, n = row & 1023;
          int h = gc >> 6, d = gc & 63;
          oq[((size_t)((b * 8 + h) * 1024 + n)) * 64 + d] = to_bf16(v * 0.125f);
        } else if (MODE == MODE_KV) {
          int b = row >> 10, n = row & 1023;
          if (gc < 512) {  // K: [b,h,m,d]
            int h = gc >> 6, d = gc & 63;
            ok[((size_t)((b * 8 + h) * 1024 + n)) * 64 + d] = to_bf16(v);
          } else {         // V transposed: [b,h,d,m]
            int c2 = gc - 512, h = c2 >> 6, d = c2 & 63;
            ov[((size_t)((b * 8 + h) * 64 + d)) * 1024 + n] = to_bf16(v);
          }
        } else {
          of[(size_t)row * 512 + gc] = v + bias[gc];
        }
      }
    }
  }
}

// ---------------------------------------------------------------------------
// Flash attention. One wave = one 16-row Q tile; 32 keys per iteration.
// K/V tiles staged to double-buffered LDS by the Tensor Data Mover (wave 0
// issues; TENSORcnt + one workgroup barrier per iteration orders producers
// and consumers). TDM pad_enable gives conflict-free LDS row strides:
//   K rows 128B -> 144B (72 bf16), V rows 64B -> 80B (40 bf16).
// ---------------------------------------------------------------------------
__global__ __launch_bounds__(256) void attn_wmma_kernel(
    const bf16* __restrict__ Q, const bf16* __restrict__ K,
    const bf16* __restrict__ Vt, bf16* __restrict__ O) {
  __shared__ bf16 Plds[8][16 * 40];  // per-wave 16x32 P patch
#if USE_TDM
  __shared__ bf16 Ktile[2][32 * 72];  // [key][d], padded rows
  __shared__ bf16 Vtile[2][64 * 40];  // [d][key], padded rows
#endif
  const int tid = threadIdx.x, wave = tid >> 5, lane = tid & 31;
  const int half = lane >> 4, l16 = lane & 15;
  const int bh = blockIdx.x >> 3;                     // b*8 + h
  const int q0 = (blockIdx.x & 7) * 128 + wave * 16;  // Q row tile base
  const bf16* Qp = Q + ((size_t)bh * 1024 + q0) * 64;
  const bf16* Kp = K + (size_t)bh * 1024 * 64;
  const bf16* Vp = Vt + (size_t)bh * 64 * 1024;
  bf16* Pw = &Plds[wave][0];

  // resident Q fragments (A 16x32 layout), two k-steps over d=64
  v16bf qf[2];
#pragma unroll
  for (int ks = 0; ks < 2; ++ks) {
    const bf16* p = Qp + (size_t)l16 * 64 + ks * 32 + half * 8;
    qf[ks] = combine16(ld8(p), ld8(p + 16));
  }

  float rmax[8], rsum[8];
#pragma unroll
  for (int r = 0; r < 8; ++r) { rmax[r] = -3.0e38f; rsum[r] = 0.f; }
  v8f acco[4] = {};

#if USE_TDM
  if (wave == 0) {  // prefetch first K/V tiles into buffer 0
    tdm_load((unsigned)(uintptr_t)&Ktile[0][0], Kp,
             0x07110000u, 0x00400000u, 0x04000000u, 0x00400000u, 32u, 64u);
    tdm_load((unsigned)(uintptr_t)&Vtile[0][0], Vp,
             0x06D10000u, 0x04000000u, 0x00400000u, 0x00200000u, 64u, 1024u);
  }
#endif
  int buf = 0;

  for (int j = 0; j < 1024; j += 32) {
#if USE_TDM
    if (wave == 0) __builtin_amdgcn_s_wait_tensorcnt(0);
    __syncthreads();  // publishes tile[buf]; proves tile[buf^1] drained
    if (wave == 0 && j + 32 < 1024) {
      tdm_load((unsigned)(uintptr_t)&Ktile[buf ^ 1][0], Kp + (size_t)(j + 32) * 64,
               0x07110000u, 0x00400000u, 0x04000000u, 0x00400000u, 32u, 64u);
      tdm_load((unsigned)(uintptr_t)&Vtile[buf ^ 1][0], Vp + (j + 32),
               0x06D10000u, 0x04000000u, 0x00400000u, 0x00200000u, 64u, 1024u);
    }
#endif

    // ---- S = Q * K^T for 32 keys ----
    v8f accs[2] = {};
#pragma unroll
    for (int nt = 0; nt < 2; ++nt) {
#pragma unroll
      for (int ks = 0; ks < 2; ++ks) {
#if USE_TDM
        const bf16* p = &Ktile[buf][(nt * 16 + l16) * 72 + ks * 32 + half * 8];
#else
        const bf16* p = Kp + (size_t)(j + nt * 16 + l16) * 64 + ks * 32 + half * 8;
#endif
        v16bf kf = combine16(ld8(p), ld8(p + 16));
        accs[nt] = wmma_bf16(qf[ks], kf, accs[nt]);
      }
    }

    // ---- online softmax (row m = r + 8*half lives across 16 lanes) ----
    float corr[8];
#pragma unroll
    for (int r = 0; r < 8; ++r) {
      float t = redmax16(fmaxf(accs[0][r], accs[1][r]));
      float nm = fmaxf(rmax[r], t);
      corr[r] = __expf(rmax[r] - nm);
      rmax[r] = nm;
    }
#pragma unroll
    for (int nt = 0; nt < 2; ++nt)
#pragma unroll
      for (int r = 0; r < 8; ++r)
        accs[nt][r] = __expf(accs[nt][r] - rmax[r]);
#pragma unroll
    for (int r = 0; r < 8; ++r) {
      float ps = redsum16(accs[0][r] + accs[1][r]);
      rsum[r] = rsum[r] * corr[r] + ps;
    }
#pragma unroll
    for (int t = 0; t < 4; ++t)
#pragma unroll
      for (int r = 0; r < 8; ++r) acco[t][r] *= corr[r];

    // ---- transpose P (C/D layout -> A layout) via per-wave LDS patch ----
#pragma unroll
    for (int nt = 0; nt < 2; ++nt)
#pragma unroll
      for (int r = 0; r < 8; ++r)
        Pw[(r + half * 8) * 40 + nt * 16 + l16] = to_bf16(accs[nt][r]);
    const bf16* pp = Pw + (size_t)l16 * 40 + half * 8;
    v16bf pf = combine16(ld8(pp), ld8(pp + 16));  // DS ops in-order per wave

    // ---- O += P * V ----
#pragma unroll
    for (int t = 0; t < 4; ++t) {
#if USE_TDM
      const bf16* p = &Vtile[buf][(t * 16 + l16) * 40 + half * 8];
#else
      const bf16* p = Vp + (size_t)(t * 16 + l16) * 1024 + j + half * 8;
#endif
      v16bf vf = combine16(ld8(p), ld8(p + 16));
      acco[t] = wmma_bf16(pf, vf, acco[t]);
    }
    buf ^= 1;
  }

  // ---- normalize, store O bf16 in [b, n, h*64+d] for the final GEMM ----
  const int b = bh >> 3, h = bh & 7;
#pragma unroll
  for (int t = 0; t < 4; ++t) {
#pragma unroll
    for (int r = 0; r < 8; ++r) {
      int n   = q0 + r + half * 8;
      int col = h * 64 + t * 16 + l16;
      O[((size_t)(b * 1024 + n)) * 512 + col] = to_bf16(acco[t][r] / rsum[r]);
    }
  }
}

// ---------------------------------------------------------------------------
extern "C" void kernel_launch(void* const* d_in, const int* in_sizes, int n_in,
                              void* d_out, int out_size, void* d_ws, size_t ws_size,
                              hipStream_t stream) {
  const float* x   = (const float*)d_in[0];  // [8,1024,512]
  const float* ctx = (const float*)d_in[1];  // [8,1024,512]
  const float* Wq  = (const float*)d_in[2];  // [512,512]
  const float* Wkv = (const float*)d_in[3];  // [512,1024]
  const float* Wo  = (const float*)d_in[4];  // [512,512]
  const float* bo  = (const float*)d_in[5];  // [512]
  float* out = (float*)d_out;

  const size_t QKV = (size_t)8 * 8 * 1024 * 64;  // 4M bf16 elems each
  bf16* Qb  = (bf16*)d_ws;
  bf16* Kb  = Qb + QKV;
  bf16* Vtb = Kb + QKV;
  bf16* Ob  = Vtb + QKV;  // [8192, 512] bf16; total ws use = 32 MB

  dim3 blk(256);
  gemm_wmma_kernel<MODE_Q, float><<<dim3(64, 4), blk, 0, stream>>>(
      x, Wq, nullptr, Qb, nullptr, nullptr, nullptr, 512, 512);
  gemm_wmma_kernel<MODE_KV, float><<<dim3(64, 8), blk, 0, stream>>>(
      ctx, Wkv, nullptr, nullptr, Kb, Vtb, nullptr, 512, 1024);
  attn_wmma_kernel<<<dim3(512), blk, 0, stream>>>(Qb, Kb, Vtb, Ob);
  gemm_wmma_kernel<MODE_OUT, bf16><<<dim3(64, 4), blk, 0, stream>>>(
      Ob, Wo, bo, nullptr, nullptr, nullptr, out, 512, 512);
}
